// ReadoutLayer_20332375180114
// MI455X (gfx1250) — compile-verified, matrix operand order checked by reading, not run on hardware
//
#include <hip/hip_runtime.h>
#include <hip/hip_bf16.h>
#include <math.h>

// Problem dims (match reference setup_inputs)
#define RB   64          // batch
#define RT   2048        // time
#define RI   1024        // input dim (GEMM K)
#define RH   512         // hidden  (GEMM N)
#define RM   (RB * RT)   // GEMM M = 131072
#define NCH  32          // chunks over T
#define CHL  (RT / NCH)  // chunk length = 64
#define KSTEPS (RI / 32) // 32 K-iterations

#define ALPHA_MIN 0.81873075307798182f
#define ALPHA_MAX 0.96078943915232320f
#define LN_EPS    1e-5f

#define LDS_W 40         // padded row width in bf16 (80 B: rows 0..15 hit distinct banks)

typedef __attribute__((ext_vector_type(16))) __bf16 v16bf;
typedef __attribute__((ext_vector_type(8)))  float  v8f;

// ---------------------------------------------------------------------------
// helpers
// ---------------------------------------------------------------------------
__device__ __forceinline__ void load16f(const float* __restrict__ p, float* r) {
    float4 a = *(const float4*)(p);
    float4 b = *(const float4*)(p + 4);
    float4 c = *(const float4*)(p + 8);
    float4 d = *(const float4*)(p + 12);
    r[0]=a.x; r[1]=a.y; r[2]=a.z;  r[3]=a.w;
    r[4]=b.x; r[5]=b.y; r[6]=b.z;  r[7]=b.w;
    r[8]=c.x; r[9]=c.y; r[10]=c.z; r[11]=c.w;
    r[12]=d.x; r[13]=d.y; r[14]=d.z; r[15]=d.w;
}

__device__ __forceinline__ void cvt_store16(const float* r, __bf16* s) {
    union { __bf16 h[16]; uint4 q[2]; } pk;
#pragma unroll
    for (int i = 0; i < 16; ++i) pk.h[i] = (__bf16)r[i];
    *(uint4*)(s)     = pk.q[0];
    *(uint4*)(s + 8) = pk.q[1];
}

// 16x32 bf16 operand fragment: lane L<16 -> row L,  K {kb..kb+7, kb+16..kb+23}
//                              lane L>=16 -> row L-16, kb += 8
// p points at lds row element [row][kOff]
__device__ __forceinline__ v16bf frag_from_lds(const __bf16* p) {
    union { v16bf v; uint4 q[2]; } f;
    f.q[0] = *(const uint4*)(p);
    f.q[1] = *(const uint4*)(p + 16);
    return f.v;
}

__device__ __forceinline__ float clamp_alpha(float a) {
    return fminf(fmaxf(a, ALPHA_MIN), ALPHA_MAX);
}

// ---------------------------------------------------------------------------
// K1: C[M,H] = x[M,I] * W[H,I]^T + b   via v_wmma_f32_16x16x32_bf16
// block = 256 threads (8 waves, 4Mx2N), block tile 128x128, wave tile 32x64.
// f32 -> bf16 conversion done once per block through double-buffered LDS.
// grid = (H/128, M/128)
// ---------------------------------------------------------------------------
__global__ __launch_bounds__(256) void gemm_bias_kernel(
    const float* __restrict__ x, const float* __restrict__ W,
    const float* __restrict__ bias, float* __restrict__ out) {
    __shared__ alignas(16) __bf16 ldsA[2][128][LDS_W];   // 10 KB per buffer
    __shared__ alignas(16) __bf16 ldsB[2][128][LDS_W];

    const int tid  = threadIdx.x;
    const int lane = tid & 31;
    const int wave = tid >> 5;
    const int mw = wave >> 1;                 // 0..3
    const int nw = wave & 1;                  // 0..1
    const int mBlock = blockIdx.y * 128;
    const int nBlock = blockIdx.x * 128;
    const int r    = lane & 15;
    const int kOff = (lane >> 4) << 3;        // 0 or 8

    // staging assignment: each thread owns 16 consecutive f32 of one slab row
    const int srow  = tid >> 1;               // 0..127
    const int shalf = (tid & 1) * 16;         // 0 or 16
    const float* gA = x + (size_t)(mBlock + srow) * RI + shalf;
    const float* gB = W + (size_t)(nBlock + srow) * RI + shalf;

    float ra[16], rb[16];
    load16f(gA, ra);
    load16f(gB, rb);
    cvt_store16(ra, &ldsA[0][srow][shalf]);
    cvt_store16(rb, &ldsB[0][srow][shalf]);
    __syncthreads();

    v8f acc[2][4] = {};

    for (int k = 0; k < KSTEPS; ++k) {
        const int buf = k & 1;
        if (k + 1 < KSTEPS) {                 // prefetch next f32 slab
            load16f(gA + (k + 1) * 32, ra);
            load16f(gB + (k + 1) * 32, rb);
        }

        v16bf af0 = frag_from_lds(&ldsA[buf][mw * 32 + r][kOff]);
        v16bf af1 = frag_from_lds(&ldsA[buf][mw * 32 + 16 + r][kOff]);
#pragma unroll
        for (int j = 0; j < 4; ++j) {
            v16bf bfj = frag_from_lds(&ldsB[buf][nw * 64 + 16 * j + r][kOff]);
            acc[0][j] = __builtin_amdgcn_wmma_f32_16x16x32_bf16(
                false, af0, false, bfj, (short)0, acc[0][j], false, false);
            acc[1][j] = __builtin_amdgcn_wmma_f32_16x16x32_bf16(
                false, af1, false, bfj, (short)0, acc[1][j], false, false);
        }

        if (k + 1 < KSTEPS) {
            __syncthreads();                  // all waves done reading buf^1
            cvt_store16(ra, &ldsA[buf ^ 1][srow][shalf]);
            cvt_store16(rb, &ldsB[buf ^ 1][srow][shalf]);
            __syncthreads();
        }
    }

    // C/D layout: lanes 0-15 -> N=lane,    M=v ; lanes 16-31 -> N=lane-16, M=v+8
    const int cOff = lane & 15;
    const int rOff = (lane >> 4) << 3;
#pragma unroll
    for (int i = 0; i < 2; ++i) {
#pragma unroll
        for (int j = 0; j < 4; ++j) {
            const int col = nBlock + nw * 64 + 16 * j + cOff;
            const float bv = bias[col];
#pragma unroll
            for (int v = 0; v < 8; ++v) {
                const int row = mBlock + mw * 32 + 16 * i + rOff + v;
                out[(size_t)row * RH + col] = acc[i][j][v] + bv;
            }
        }
    }
}

// ---------------------------------------------------------------------------
// K2: per-row LayerNorm statistics only: stats[row] = {mu, rstd}. 1 wave/row.
// ---------------------------------------------------------------------------
__global__ __launch_bounds__(256) void ln_stats_kernel(
    const float* __restrict__ wx, float2* __restrict__ stats) {
    const int lane = threadIdx.x & 31;
    const int wave = threadIdx.x >> 5;
    const size_t row = (size_t)blockIdx.x * 8 + wave;
    const float* p = wx + row * RH;

    float s = 0.f, s2 = 0.f;
#pragma unroll
    for (int j = 0; j < 16; ++j) {
        float v = p[j * 32 + lane];
        s  += v;
        s2 += v * v;
    }
#pragma unroll
    for (int off = 16; off > 0; off >>= 1) {
        s  += __shfl_xor(s,  off, 32);
        s2 += __shfl_xor(s2, off, 32);
    }
    if (lane == 0) {
        const float mu  = s * (1.f / RH);
        const float var = s2 * (1.f / RH) - mu * mu;
        stats[row] = make_float2(mu, rsqrtf(var + LN_EPS));
    }
}

// ---------------------------------------------------------------------------
// K3: per-chunk local scan with on-the-fly LN:
//     v_c = sum_i a^(L-1-i) (1-a) wn_i   (u_start = 0)
// grid (NCH, B), block H threads.
// ---------------------------------------------------------------------------
__global__ __launch_bounds__(512) void chunk_scan_kernel(
    const float* __restrict__ wx, const float2* __restrict__ stats,
    const float* __restrict__ alpha, const float* __restrict__ ln_g,
    const float* __restrict__ ln_b, float* __restrict__ vchunk) {
    const int h = threadIdx.x;
    const int c = blockIdx.x;
    const int b = blockIdx.y;
    const float a  = clamp_alpha(alpha[h]);
    const float oa = 1.f - a;
    const float gh = ln_g[h];
    const float bh = ln_b[h];
    const size_t rowbase = (size_t)b * RT + (size_t)c * CHL;
    const float* p = wx + rowbase * RH + h;
    float v = 0.f;
    for (int i = 0; i < CHL; ++i) {
        const float2 st = stats[rowbase + i];          // uniform -> scalar load
        const float wn = (p[(size_t)i * RH] - st.x) * st.y * gh + bh;
        v = a * v + oa * wn;
    }
    vchunk[((size_t)b * NCH + c) * RH + h] = v;
}

// ---------------------------------------------------------------------------
// K4: sequential prefix over NCH chunk summaries -> u_start per chunk.
// ---------------------------------------------------------------------------
__global__ __launch_bounds__(512) void chunk_prefix_kernel(
    const float* __restrict__ vchunk, const float* __restrict__ alpha,
    const float* __restrict__ ut0, float* __restrict__ ustart) {
    const int h = threadIdx.x;
    const int b = blockIdx.x;
    const float a = clamp_alpha(alpha[h]);
    float aL = 1.f;
#pragma unroll
    for (int i = 0; i < CHL; ++i) aL *= a;             // a^CHL
    float u = ut0[(size_t)b * RH + h];
    for (int c = 0; c < NCH; ++c) {
        const size_t idx = ((size_t)b * NCH + c) * RH + h;
        ustart[idx] = u;
        u = aL * u + vchunk[idx];
    }
}

// ---------------------------------------------------------------------------
// K5: replay each chunk: LN + u update + softmax over H, accumulate partials.
// grid (NCH, B), block H=512 (16 waves). Loads pipelined past the reductions.
// ---------------------------------------------------------------------------
__global__ __launch_bounds__(512) void softmax_acc_kernel(
    const float* __restrict__ wx, const float2* __restrict__ stats,
    const float* __restrict__ alpha, const float* __restrict__ ln_g,
    const float* __restrict__ ln_b, const float* __restrict__ ustart,
    float* __restrict__ partial) {
    __shared__ float redm[16];
    __shared__ float reds[16];
    const int h = threadIdx.x;
    const int lane = h & 31;
    const int wid  = h >> 5;
    const int c = blockIdx.x;
    const int b = blockIdx.y;
    const float a  = clamp_alpha(alpha[h]);
    const float oa = 1.f - a;
    const float gh = ln_g[h];
    const float bh = ln_b[h];

    float u = ustart[((size_t)b * NCH + c) * RH + h];
    const size_t rowbase = (size_t)b * RT + (size_t)c * CHL;
    const float* p = wx + rowbase * RH + h;
    float acc = 0.f;

    float  wcur = p[0];
    float2 stc  = stats[rowbase];

    for (int i = 0; i < CHL; ++i) {
        const float wn = (wcur - stc.x) * stc.y * gh + bh;
        if (i + 1 < CHL) {                    // issue next loads before barriers
            wcur = p[(size_t)(i + 1) * RH];
            stc  = stats[rowbase + i + 1];
        }
        u = a * u + oa * wn;

        // block max over H
        float m = u;
#pragma unroll
        for (int off = 16; off > 0; off >>= 1) m = fmaxf(m, __shfl_xor(m, off, 32));
        if (lane == 0) redm[wid] = m;
        __syncthreads();
        m = redm[0];
#pragma unroll
        for (int w = 1; w < 16; ++w) m = fmaxf(m, redm[w]);

        const float e = __expf(u - m);

        // block sum over H
        float s = e;
#pragma unroll
        for (int off = 16; off > 0; off >>= 1) s += __shfl_xor(s, off, 32);
        if (lane == 0) reds[wid] = s;
        __syncthreads();
        s = reds[0];
#pragma unroll
        for (int w = 1; w < 16; ++w) s += reds[w];

        acc += e / s;
        __syncthreads();                      // protect redm/reds for next iter
    }
    partial[((size_t)b * NCH + c) * RH + h] = acc;
}

// ---------------------------------------------------------------------------
// K6: deterministic reduction of chunk partials -> out[B,H]
// ---------------------------------------------------------------------------
__global__ __launch_bounds__(256) void final_reduce_kernel(
    const float* __restrict__ partial, float* __restrict__ out) {
    const int idx = blockIdx.x * 256 + threadIdx.x;    // 0 .. B*H-1
    const int b = idx / RH;
    const int h = idx - b * RH;
    float s = 0.f;
    for (int c = 0; c < NCH; ++c)
        s += partial[((size_t)b * NCH + c) * RH + h];
    out[idx] = s;
}

// ---------------------------------------------------------------------------
extern "C" void kernel_launch(void* const* d_in, const int* in_sizes, int n_in,
                              void* d_out, int out_size, void* d_ws, size_t ws_size,
                              hipStream_t stream) {
    (void)in_sizes; (void)n_in; (void)out_size; (void)ws_size;
    const float* x     = (const float*)d_in[0];
    const float* W     = (const float*)d_in[1];
    const float* bias  = (const float*)d_in[2];
    const float* alpha = (const float*)d_in[3];
    const float* ln_g  = (const float*)d_in[4];
    const float* ln_b  = (const float*)d_in[5];
    const float* ut0   = (const float*)d_in[6];
    float* out = (float*)d_out;

    char* ws = (char*)d_ws;
    const size_t WX_BYTES = (size_t)RM * RH * sizeof(float);         // 256 MB
    const size_t ST_BYTES = (size_t)RM * sizeof(float2);             // 1 MB
    const size_t CK_BYTES = (size_t)RB * NCH * RH * sizeof(float);   // 4 MB
    float*  wx      = (float*)(ws);
    float2* stats   = (float2*)(ws + WX_BYTES);
    float*  vchunk  = (float*)(ws + WX_BYTES + ST_BYTES);
    float*  ustart  = (float*)(ws + WX_BYTES + ST_BYTES + CK_BYTES);
    float*  partial = (float*)(ws + WX_BYTES + ST_BYTES + 2 * CK_BYTES);

    dim3 g1(RH / 128, RM / 128);                      // (4, 1024)
    gemm_bias_kernel<<<g1, 256, 0, stream>>>(x, W, bias, wx);

    ln_stats_kernel<<<RM / 8, 256, 0, stream>>>(wx, stats);

    dim3 g3(NCH, RB);                                 // (32, 64)
    chunk_scan_kernel<<<g3, RH, 0, stream>>>(wx, stats, alpha, ln_g, ln_b, vchunk);

    chunk_prefix_kernel<<<RB, RH, 0, stream>>>(vchunk, alpha, ut0, ustart);

    dim3 g5(NCH, RB);
    softmax_acc_kernel<<<g5, RH, 0, stream>>>(wx, stats, alpha, ln_g, ln_b,
                                              ustart, partial);

    final_reduce_kernel<<<(RB * RH) / 256, 256, 0, stream>>>(partial, out);
}
